// Encoder_35450660061949
// MI455X (gfx1250) — matrix operand order repaired and checked by reading, not compiled, not run on hardware
//
#include <hip/hip_runtime.h>
#include <hip/hip_bf16.h>

// ---------------------------------------------------------------------------
// 2-layer GAT for MI455X (gfx1250).
//   layer: h = x@W (WMMA f16->f32), per-node attention scores, per-edge
//   leaky-relu + segment softmax over dst, weighted scatter-add, bias+relu.
// Dense GEMMs use v_wmma_f32_16x16x32_f16 (wave32, 16x16 tile per wave).
// Edge passes are wave/lane-parallel with f32 global atomics (all node arrays
// fit in the 192MB L2, so scattered traffic stays on-chip).
// ---------------------------------------------------------------------------

typedef __attribute__((ext_vector_type(16))) _Float16 v16h;
typedef __attribute__((ext_vector_type(8)))  _Float16 v8h;
typedef __attribute__((ext_vector_type(8)))  float    v8f;

#define NEG_SLOPE 0.2f

// ---------------- monotonic float<->int ordering for atomicMax -------------
__device__ __forceinline__ int f2ord(float f) {
    int i = __float_as_int(f);
    return i >= 0 ? i : (i ^ 0x7fffffff);
}
__device__ __forceinline__ float ord2f(int v) {
    int i = v >= 0 ? v : (v ^ 0x7fffffff);
    return __int_as_float(i);
}
// f2ord(-inf) == 0x807FFFFF
#define MORD_NEG_INF ((int)0x807FFFFF)

// ---------------- fills ----------------------------------------------------
__global__ void fill_f32(float* __restrict__ p, long long n, float v) {
    long long i = (long long)blockIdx.x * blockDim.x + threadIdx.x;
    if (i < n) p[i] = v;
}
__global__ void fill_i32(int* __restrict__ p, long long n, int v) {
    long long i = (long long)blockIdx.x * blockDim.x + threadIdx.x;
    if (i < n) p[i] = v;
}

// ---------------- conversions / padding ------------------------------------
// x [N,100] f32 -> xh [N,128] f16 (zero pad K)
__global__ void cvt_x_pad(const float* __restrict__ x, _Float16* __restrict__ xh,
                          long long N) {
    long long i = (long long)blockIdx.x * blockDim.x + threadIdx.x;
    if (i >= N * 128) return;
    long long r = i >> 7;
    int c = (int)(i & 127);
    xh[i] = (_Float16)(c < 100 ? x[r * 100 + c] : 0.0f);
}

// W [KIN,NIN] f32 -> Wt [NP,KP] f16 transposed + zero padded
template <int KIN, int NIN, int KP, int NP>
__global__ void cvt_w(const float* __restrict__ W, _Float16* __restrict__ Wt) {
    int i = blockIdx.x * blockDim.x + threadIdx.x;
    if (i >= NP * KP) return;
    int n = i / KP, k = i % KP;
    float v = (n < NIN && k < KIN) ? W[k * NIN + n] : 0.0f;
    Wt[i] = (_Float16)v;
}

// ---------------- WMMA GEMM: C[M,NT*16] = A[M,KP] * Bt[NT*16,KP]^T ---------
// One wave per 16x16 output tile. A row-major f16 (KP multiple of 32),
// Bt is B transposed (row n holds column n of B, contiguous in k).
// Fragment layouts per CDNA5 ISA 7.12.2:
//   A 16x32: lane m=l&15, halves: k = 8*(l>>4)+0..7 then 16+8*(l>>4)+0..7
//   B 32x16: lane n=l&15, halves: k = 8*(l>>4)+0..7 then 16+8*(l>>4)+0..7
//   C 16x16: lane n=l&15, vgpr v: m = v + 8*(l>>4)
template <int KP, int NT>
__global__ __launch_bounds__(256) void wmma_gemm(
    const _Float16* __restrict__ A, const _Float16* __restrict__ Bt,
    float* __restrict__ C, int Mtiles) {
    int wave = blockIdx.x * (blockDim.x >> 5) + (threadIdx.x >> 5);
    int lane = threadIdx.x & 31;
    if (wave >= Mtiles * NT) return;
    int tm = wave / NT, tn = wave % NT;
    int half = lane >> 4, l16 = lane & 15;

    const _Float16* arow = A + (size_t)(tm * 16 + l16) * KP + 8 * half;
    const _Float16* brow = Bt + (size_t)(tn * 16 + l16) * KP + 8 * half;

    v8f acc = {};
#pragma unroll
    for (int k0 = 0; k0 < KP; k0 += 32) {
        v8h alo = *(const v8h*)(arow + k0);
        v8h ahi = *(const v8h*)(arow + k0 + 16);
        v16h a;
#pragma unroll
        for (int i = 0; i < 8; ++i) { a[i] = alo[i]; a[i + 8] = ahi[i]; }
        v8h blo = *(const v8h*)(brow + k0);
        v8h bhi = *(const v8h*)(brow + k0 + 16);
        v16h b;
#pragma unroll
        for (int i = 0; i < 8; ++i) { b[i] = blo[i]; b[i + 8] = bhi[i]; }
        acc = __builtin_amdgcn_wmma_f32_16x16x32_f16(
            false, a, false, b, (short)0, acc, false, false);
    }
    const int ldc = NT * 16;
    float* cp = C + (size_t)(tm * 16 + 8 * half) * ldc + tn * 16 + l16;
#pragma unroll
    for (int v = 0; v < 8; ++v) cp[(size_t)v * ldc] = acc[v];
}

// ---------------- per-node attention halves --------------------------------
__global__ void node_scores(const float* __restrict__ h,
                            const float* __restrict__ asrc,
                            const float* __restrict__ adst,
                            float* __restrict__ ssrc, float* __restrict__ sdst,
                            long long N, int ld, int F) {
    long long i = (long long)blockIdx.x * blockDim.x + threadIdx.x;
    if (i >= N) return;
    const float* row = h + (size_t)i * ld;
    float a = 0.f, b = 0.f;
    for (int f = 0; f < F; ++f) {
        float v = row[f];
        a += v * asrc[f];
        b += v * adst[f];
    }
    ssrc[i] = a;
    sdst[i] = b;
}

// ---------------- edge helpers (implicit self-loops at i>=E) ---------------
__device__ __forceinline__ void edge_sd(const long long* __restrict__ ei,
                                        long long E, long long i, int& s, int& d) {
    if (i < E) { s = (int)ei[i]; d = (int)ei[E + i]; }
    else       { s = d = (int)(i - E); }
}

__global__ void edge_max(const long long* __restrict__ ei, long long E,
                         long long Etot, const float* __restrict__ ss,
                         const float* __restrict__ sd, int* __restrict__ mord) {
    long long i = (long long)blockIdx.x * blockDim.x + threadIdx.x;
    if (i >= Etot) return;
    int s, d;
    edge_sd(ei, E, i, s, d);
    float a = ss[s] + sd[d];
    a = a > 0.f ? a : NEG_SLOPE * a;
    atomicMax(&mord[d], f2ord(a));
}

__global__ void edge_expsum(const long long* __restrict__ ei, long long E,
                            long long Etot, const float* __restrict__ ss,
                            const float* __restrict__ sd,
                            const int* __restrict__ mord,
                            float* __restrict__ denom,
                            float* __restrict__ evals) {
    long long i = (long long)blockIdx.x * blockDim.x + threadIdx.x;
    if (i >= Etot) return;
    int s, d;
    edge_sd(ei, E, i, s, d);
    float a = ss[s] + sd[d];
    a = a > 0.f ? a : NEG_SLOPE * a;
    float ev = __expf(a - ord2f(mord[d]));
    evals[i] = ev;
    atomicAdd(&denom[d], ev);
}

// ---------------- weighted scatter-add of source messages ------------------
// LPE lanes cooperate on one edge (coalesced gather + coalesced atomics).
template <int F, int LDH, int LDO, int LPE>
__global__ void aggregate(const long long* __restrict__ ei, long long E,
                          long long Etot, const float* __restrict__ h,
                          const float* __restrict__ evals,
                          const float* __restrict__ denom,
                          float* __restrict__ out) {
    const int epw = 32 / LPE;  // edges per wave
    int lane = threadIdx.x & 31;
    long long edge =
        ((long long)blockIdx.x * (blockDim.x >> 5) + (threadIdx.x >> 5)) * epw +
        (lane / LPE);
    int fl = lane % LPE;
    if (edge >= Etot) return;
    int s, d;
    edge_sd(ei, E, edge, s, d);
    float coef = evals[edge] / (denom[d] + 1e-16f);
    const float* hs = h + (size_t)s * LDH;
    float* od = out + (size_t)d * LDO;
#pragma unroll
    for (int f = fl; f < F; f += LPE) atomicAdd(&od[f], hs[f] * coef);
}

// ---------------- bias + relu epilogues ------------------------------------
// layer1: acc [N,64] (50 valid) -> relu(acc+b) as f16 [N,64] (pad zeroed)
__global__ void bias_relu_to_f16(const float* __restrict__ acc,
                                 const float* __restrict__ b,
                                 _Float16* __restrict__ out, long long N) {
    long long i = (long long)blockIdx.x * blockDim.x + threadIdx.x;
    if (i >= N * 64) return;
    int c = (int)(i & 63);
    float v = 0.f;
    if (c < 50) {
        v = acc[i] + b[c];
        v = v > 0.f ? v : 0.f;
    }
    out[i] = (_Float16)v;
}

// layer2: acc [N,4] -> relu(acc+b) f32 (final output)
__global__ void bias_relu_out(const float* __restrict__ acc,
                              const float* __restrict__ b,
                              float* __restrict__ out, long long N) {
    long long i = (long long)blockIdx.x * blockDim.x + threadIdx.x;
    if (i >= N * 4) return;
    float v = acc[i] + b[i & 3];
    out[i] = v > 0.f ? v : 0.f;
}

// ---------------------------------------------------------------------------
extern "C" void kernel_launch(void* const* d_in, const int* in_sizes, int n_in,
                              void* d_out, int out_size, void* d_ws,
                              size_t ws_size, hipStream_t stream) {
    (void)n_in; (void)out_size; (void)ws_size;
    const float*     x   = (const float*)d_in[0];
    const long long* ei  = (const long long*)d_in[1];  // int64 [2,E]
    const float*     W1  = (const float*)d_in[2];
    const float*     as1 = (const float*)d_in[3];
    const float*     ad1 = (const float*)d_in[4];
    const float*     b1  = (const float*)d_in[5];
    const float*     W2  = (const float*)d_in[6];
    const float*     as2 = (const float*)d_in[7];
    const float*     ad2 = (const float*)d_in[8];
    const float*     b2  = (const float*)d_in[9];

    const long long N    = in_sizes[0] / 100;  // 100000 (multiple of 16)
    const long long E    = in_sizes[1] / 2;    // 1600000
    const long long Etot = E + N;              // + self loops
    const int Mtiles = (int)(N / 16);

    // ---- workspace layout (with lifetime-based aliasing) ----
    char* ws = (char*)d_ws;
    size_t off = 0;
    auto alloc = [&](size_t bytes) -> void* {
        void* p = ws + off;
        off += (bytes + 255) & ~(size_t)255;
        return p;
    };
    _Float16* xh   = (_Float16*)alloc((size_t)N * 128 * 2);  // region A (25.6MB)
    float*    agg1 = (float*)xh;            // alias A: xh dead after GEMM1
    float*    h1   = (float*)alloc((size_t)N * 64 * 4);      // region B (25.6MB)
    _Float16* h1r  = (_Float16*)h1;         // alias B lower half: h1 dead after aggregate1
    float*    h2   = (float*)((char*)h1 + (size_t)N * 64 * 2); // B upper half
    _Float16* W1t  = (_Float16*)alloc(64 * 128 * 2);
    _Float16* W2t  = (_Float16*)alloc(16 * 64 * 2);
    float* ssrc  = (float*)alloc((size_t)N * 4);
    float* sdst  = (float*)alloc((size_t)N * 4);
    int*   mord  = (int*)alloc((size_t)N * 4);
    float* denom = (float*)alloc((size_t)N * 4);
    float* evals = (float*)alloc((size_t)Etot * 4);
    float* agg2  = (float*)alloc((size_t)N * 4 * 4);

    auto nb = [](long long n) { return (unsigned)((n + 255) / 256); };

    // ================= layer 1 =================
    cvt_x_pad<<<nb(N * 128), 256, 0, stream>>>(x, xh, N);
    cvt_w<100, 50, 128, 64><<<nb(64 * 128), 256, 0, stream>>>(W1, W1t);
    cvt_w<50, 4, 64, 16><<<nb(16 * 64), 256, 0, stream>>>(W2, W2t);

    // h1 = x @ W1   (6250*4 tiles, 8 waves/block)
    wmma_gemm<128, 4><<<(unsigned)((Mtiles * 4 + 7) / 8), 256, 0, stream>>>(
        xh, W1t, h1, Mtiles);

    node_scores<<<nb(N), 256, 0, stream>>>(h1, as1, ad1, ssrc, sdst, N, 64, 50);

    fill_i32<<<nb(N), 256, 0, stream>>>(mord, N, MORD_NEG_INF);
    fill_f32<<<nb(N), 256, 0, stream>>>(denom, N, 0.f);
    fill_f32<<<nb(N * 64), 256, 0, stream>>>(agg1, N * 64, 0.f);

    edge_max<<<nb(Etot), 256, 0, stream>>>(ei, E, Etot, ssrc, sdst, mord);
    edge_expsum<<<nb(Etot), 256, 0, stream>>>(ei, E, Etot, ssrc, sdst, mord,
                                              denom, evals);
    aggregate<50, 64, 64, 32><<<nb(Etot * 32), 256, 0, stream>>>(
        ei, E, Etot, h1, evals, denom, agg1);

    bias_relu_to_f16<<<nb(N * 64), 256, 0, stream>>>(agg1, b1, h1r, N);

    // ================= layer 2 =================
    // h2 = relu(out1) @ W2
    wmma_gemm<64, 1><<<(unsigned)((Mtiles + 7) / 8), 256, 0, stream>>>(
        h1r, W2t, h2, Mtiles);

    node_scores<<<nb(N), 256, 0, stream>>>(h2, as2, ad2, ssrc, sdst, N, 16, 4);

    fill_i32<<<nb(N), 256, 0, stream>>>(mord, N, MORD_NEG_INF);
    fill_f32<<<nb(N), 256, 0, stream>>>(denom, N, 0.f);
    fill_f32<<<nb(N * 4), 256, 0, stream>>>(agg2, N * 4, 0.f);

    edge_max<<<nb(Etot), 256, 0, stream>>>(ei, E, Etot, ssrc, sdst, mord);
    edge_expsum<<<nb(Etot), 256, 0, stream>>>(ei, E, Etot, ssrc, sdst, mord,
                                              denom, evals);
    aggregate<4, 16, 4, 4><<<nb(Etot * 4), 256, 0, stream>>>(
        ei, E, Etot, h2, evals, denom, agg2);

    bias_relu_out<<<nb(N * 4), 256, 0, stream>>>(agg2, b2, (float*)d_out, N);
}
